// AMS_55490977464462
// MI455X (gfx1250) — compile-verified, matrix operand order checked by reading, not run on hardware
//
#include <hip/hip_runtime.h>
#include <cstdint>
#include <math.h>

#define Bn 32
#define Ln 96
#define Nn 64
#define Dn 128
#define DFFn 256
#define Hn 4
#define En 4
#define HDn 32
#define LN_EPS 1e-5f

typedef __attribute__((ext_vector_type(16))) _Float16 v16h;
typedef __attribute__((ext_vector_type(8)))  float    v8f;

// ---------------------------------------------------------------------------
// Fragment layout helpers (CDNA5 ISA 7.12.2, 16-bit A 16x32 operand layout).
// A v16h element i lives in VGPR i/2 (half i%2). For lane L (row/col = L%16):
//   k = (v>=4?16:0) + (L>=16?8:0) + 2*(v&3) + half
// ---------------------------------------------------------------------------
__device__ __forceinline__ int kmap(int lane, int i) {
    int v = i >> 1, half = i & 1;
    return ((v >= 4) ? 16 : 0) + ((lane >= 16) ? 8 : 0) + ((v & 3) << 1) + half;
}
// inverse: element (r, kk) of a 16x32 fragment -> flat f16 index in a buffer
// laid out [frag = mt*fragsK + kt][lane(32)][i(16)]
__device__ __forceinline__ int frag_off(int mt, int kt, int fragsK, int r, int kk) {
    int lane = r + (((kk & 15) >= 8) ? 16 : 0);
    int v    = ((kk >= 16) ? 4 : 0) + ((kk & 7) >> 1);
    int i    = (v << 1) | (kk & 1);
    return (((mt * fragsK + kt) << 5) + lane) * 16 + i;
}

// 1x2 register-blocked tile pair: one A fragment (LDS) feeds two B tiles
// (global, pre-swizzled) -> 2 WMMAs per A load, halves LDS traffic per WMMA.
__device__ __forceinline__ void gemm_tile_x2(const _Float16* A, const _Float16* Bw,
                                             int mt, int ntp, int fragsK, int lane,
                                             v8f& acc0, v8f& acc1) {
    const v16h* Afr = (const v16h*)A;
    const v16h* Bfr = (const v16h*)Bw;
    int abase = (mt * fragsK) * 32 + lane;
    int b0 = ((2 * ntp + 0) * fragsK) * 32 + lane;
    int b1 = ((2 * ntp + 1) * fragsK) * 32 + lane;
    for (int kt = 0; kt < fragsK; ++kt) {
        v16h a  = Afr[abase + kt * 32];
        v16h bA = Bfr[b0 + kt * 32];
        v16h bB = Bfr[b1 + kt * 32];
        acc0 = __builtin_amdgcn_wmma_f32_16x16x32_f16(false, a, false, bA,
                                                      (short)0, acc0, false, false);
        acc1 = __builtin_amdgcn_wmma_f32_16x16x32_f16(false, a, false, bB,
                                                      (short)0, acc1, false, false);
    }
}

// ---------------------------------------------------------------------------
// Kernel 1: pre-swizzle expert weights (f32 row-major [NO][K]) -> f16 fragments
// per-expert region layout (f16 counts): qkv 49152 | wo 16384 | ff1 32768 | ff2 32768
// ---------------------------------------------------------------------------
__global__ __launch_bounds__(256) void wswz_kernel(
    const float* __restrict__ w_in, const float* __restrict__ w_out,
    const float* __restrict__ w_ff1, const float* __restrict__ w_ff2,
    _Float16* __restrict__ dst)
{
    int gid = blockIdx.x * 256 + threadIdx.x;  // 0 .. 4*131072-1
    int e = gid / 131072, r = gid % 131072;
    const float* src; int K, base;
    if (r < 49152)                      { src = w_in  + e * 384 * 128; K = 128; base = r; }
    else if (r < 49152 + 16384)         { src = w_out + e * 128 * 128; K = 128; base = r - 49152; }
    else if (r < 49152 + 16384 + 32768) { src = w_ff1 + e * 256 * 128; K = 128; base = r - 65536; }
    else                                { src = w_ff2 + e * 128 * 256; K = 256; base = r - 98304; }
    int frag = base >> 9, rem = base & 511;
    int lane = rem >> 4, i = rem & 15;
    int fragsK = K >> 5;
    int nt = frag / fragsK, kt = frag % fragsK;
    int nn = nt * 16 + (lane & 15);
    int k  = kt * 32 + kmap(lane, i);
    dst[gid] = (_Float16)src[nn * K + k];
}

// ---------------------------------------------------------------------------
// Kernel 2: series decomposition. One wave32 per (b,n) series of length L=96.
// new_x = x0 + fourier_season(top-3 freq) + mean of moving averages {3,7,11}
// ---------------------------------------------------------------------------
__global__ __launch_bounds__(256) void decomp_kernel(const float* __restrict__ x,
                                                     float* __restrict__ new_x)
{
    __shared__ float s_x[8][Ln];
    __shared__ float s_re[8][49], s_im[8][49], s_mag[8][49];
    __shared__ float s_th[8];
    int wv = threadIdx.x >> 5, lane = threadIdx.x & 31;
    int series = blockIdx.x * 8 + wv;           // 0..2047
    int b = series / Nn, n = series % Nn;
    const float* xp = x + ((size_t)b * Ln * Nn + n) * Dn;  // x[b,l,n,0] at l*N*D
    for (int t = lane; t < Ln; t += 32) s_x[wv][t] = xp[(size_t)t * Nn * Dn];
    __syncthreads();
    const float W0 = 6.28318530717958647f / 96.f;
    for (int f = lane; f < 49; f += 32) {
        float re = 0.f, im = 0.f, w = W0 * (float)f;
        for (int t = 0; t < Ln; ++t) {
            float s, c; __sincosf(w * (float)t, &s, &c);
            re += s_x[wv][t] * c;
            im -= s_x[wv][t] * s;
        }
        s_re[wv][f] = re; s_im[wv][f] = im;
        s_mag[wv][f] = (f == 0) ? 0.f : sqrtf(re * re + im * im);
    }
    __syncthreads();
    if (lane == 0) {  // third-largest magnitude (multiplicity-aware)
        float m1 = -1.f, m2 = -1.f, m3 = -1.f;
        for (int f = 0; f < 49; ++f) {
            float v = s_mag[wv][f];
            if (v > m1)      { m3 = m2; m2 = m1; m1 = v; }
            else if (v > m2) { m3 = m2; m2 = v; }
            else if (v > m3) { m3 = v; }
        }
        s_th[wv] = m3;
    }
    __syncthreads();
    float th = s_th[wv];
    for (int t = lane; t < Ln; t += 32) {
        float season = 0.f;
        for (int f = 0; f < 49; ++f) {
            if (s_mag[wv][f] >= th) {
                float wgt = (f == 0 || f == 48) ? 1.f : 2.f;
                float s, c; __sincosf(W0 * (float)f * (float)t, &s, &c);
                season += wgt * (s_re[wv][f] * c - s_im[wv][f] * s);
            }
        }
        season *= (1.f / 96.f);
        float trend = 0.f;
        const int ks[3] = {3, 7, 11};
        for (int ki = 0; ki < 3; ++ki) {
            int k = ks[ki], pl = (k - 1) / 2;
            float acc = 0.f;
            for (int j = 0; j < k; ++j) {
                int idx = t - pl + j;
                idx = idx < 0 ? 0 : (idx > Ln - 1 ? Ln - 1 : idx);
                acc += s_x[wv][idx];
            }
            trend += acc / (float)k;
        }
        trend *= (1.f / 3.f);
        new_x[((size_t)b * Ln + t) * Nn + n] = s_x[wv][t] + season + trend;
    }
}

// ---------------------------------------------------------------------------
// Kernel 3: gating (single workgroup) -> gates[B][E] and balance loss
// ---------------------------------------------------------------------------
__global__ __launch_bounds__(256) void gate_kernel(
    const float* __restrict__ new_x, const float* __restrict__ w_sl,
    const float* __restrict__ b_sl, const float* __restrict__ w_gate,
    const float* __restrict__ b_gate, float* __restrict__ gates,
    float* __restrict__ loss_out)
{
    __shared__ float s_gin[Bn * Ln];
    __shared__ float s_logit[Bn * En];
    __shared__ float s_g[Bn * En];
    int tid = threadIdx.x;
    for (int idx = tid; idx < Bn * Ln; idx += 256) {
        int b = idx / Ln, l = idx % Ln;
        const float* row = new_x + ((size_t)b * Ln + l) * Nn;
        float s = 0.f;
        for (int n = 0; n < Nn; ++n) s += row[n] * w_sl[n];
        s_gin[idx] = s + b_sl[0];
    }
    __syncthreads();
    for (int idx = tid; idx < Bn * En; idx += 256) {
        int b = idx / En, e = idx % En;
        float s = 0.f;
        for (int l = 0; l < Ln; ++l) s += s_gin[b * Ln + l] * w_gate[e * Ln + l];
        s_logit[idx] = s + b_gate[e];
    }
    __syncthreads();
    if (tid < Bn) {
        int b = tid;
        float v[En];
        for (int e = 0; e < En; ++e) v[e] = s_logit[b * En + e];
        int i1 = 0;
        for (int e = 1; e < En; ++e) if (v[e] > v[i1]) i1 = e;
        int i2 = -1;
        for (int e = 0; e < En; ++e) { if (e == i1) continue; if (i2 < 0 || v[e] > v[i2]) i2 = e; }
        float m  = fmaxf(v[i1], v[i2]);
        float e1 = __expf(v[i1] - m), e2 = __expf(v[i2] - m);
        float inv = 1.f / (e1 + e2);
        for (int e = 0; e < En; ++e) s_g[b * En + e] = 0.f;
        s_g[b * En + i1] = e1 * inv;
        s_g[b * En + i2] = e2 * inv;
        for (int e = 0; e < En; ++e) gates[b * En + e] = s_g[b * En + e];
    }
    __syncthreads();
    if (tid == 0) {
        float imp[En], ld[En];
        for (int e = 0; e < En; ++e) { imp[e] = 0.f; ld[e] = 0.f; }
        for (int b = 0; b < Bn; ++b)
            for (int e = 0; e < En; ++e) {
                float gv = s_g[b * En + e];
                imp[e] += gv;
                ld[e]  += (gv > 0.f) ? 1.f : 0.f;
            }
        auto cv2 = [](const float* a) {
            float mean = 0.f;
            for (int e = 0; e < En; ++e) mean += a[e];
            mean *= (1.f / En);
            float var = 0.f;
            for (int e = 0; e < En; ++e) { float d = a[e] - mean; var += d * d; }
            var *= (1.f / (En - 1));
            return var / (mean * mean + 1e-10f);
        };
        loss_out[0] = 0.01f * (cv2(imp) + cv2(ld));
    }
}

// ---------------------------------------------------------------------------
// Kernel 4: one expert over one (b,n) panel of 96 tokens x 128 features.
// 8 waves per WG; all GEMMs via v_wmma_f32_16x16x32_f16 (1x2 N-blocked).
// Dynamic LDS is declared as an array of 32-byte v16h so the base is 32-B
// aligned AND all derived pointers stay in addrspace(3) -> native ds_*.
// ---------------------------------------------------------------------------
template <int PATCH>
__global__ __launch_bounds__(256) void expert_kernel(
    const float* __restrict__ x, const float* __restrict__ gates,
    const _Float16* __restrict__ wf16,
    const float* __restrict__ b_in, const float* __restrict__ b_out,
    const float* __restrict__ ln1_g, const float* __restrict__ ln1_b,
    const float* __restrict__ b_ff1, const float* __restrict__ b_ff2,
    const float* __restrict__ ln2_g, const float* __restrict__ ln2_b,
    float* __restrict__ out, int expert, int first)
{
    extern __shared__ v16h smem_v[];               // 32-B aligned dynamic LDS
    char* sm = (char*)smem_v;
    float*    S_f32 = (float*)sm;                  // 96*128 f32  (48 KB)
    _Float16* S_big = (_Float16*)(sm + 49152);     // 96*384 f16  (72 KB)
    _Float16* S_A   = (_Float16*)(sm + 49152 + 73728); // 96*128 f16 (24 KB)
    __shared__ float s_gate;

    int wg = blockIdx.x;
    int b = wg / Nn, n = wg % Nn;
    int tid = threadIdx.x, lane = tid & 31, wv = tid >> 5;

    if (tid == 0) s_gate = gates[b * En + expert];
    __syncthreads();
    float g = s_gate;

    const float* xrow = x + ((size_t)b * Ln * Nn + n) * Dn;  // (l,d) at l*N*D + d
    float*       orow = out + ((size_t)b * Ln * Nn + n) * Dn;

    if (g == 0.0f) {                 // expert inactive: contributes 0
        if (first) {                 // first expert seeds out = x (residual)
            for (int idx = tid; idx < Ln * Dn; idx += 256) {
                int l = idx >> 7, d = idx & 127;
                orow[(size_t)l * (Nn * Dn) + d] = xrow[(size_t)l * (Nn * Dn) + d];
            }
        }
        return;
    }

    const _Float16* Wqkv = wf16;
    const _Float16* Wwo  = wf16 + 49152;
    const _Float16* Wff1 = wf16 + 49152 + 16384;
    const _Float16* Wff2 = wf16 + 49152 + 16384 + 32768;
    if (tid == 0) {  // warm L2 for weights (global_prefetch_b8)
        __builtin_prefetch(Wqkv, 0, 3);
        __builtin_prefetch(Wwo,  0, 3);
        __builtin_prefetch(Wff1, 0, 3);
        __builtin_prefetch(Wff2, 0, 3);
    }

    // Phase 1: load panel -> f32 (row-major) + f16 (fragment-swizzled A)
    for (int idx = tid; idx < Ln * Dn; idx += 256) {
        int l = idx >> 7, d = idx & 127;
        float v = xrow[(size_t)l * (Nn * Dn) + d];
        S_f32[idx] = v;
        S_A[frag_off(l >> 4, d >> 5, 4, l & 15, d & 31)] = (_Float16)v;
    }
    __syncthreads();

    // Phase 2: QKV = h @ w_in^T + b_in  (96x384, K=128) -> S_big row-major f16
    for (int t = wv; t < 6 * 12; t += 8) {
        int mt = t / 12, ntp = t % 12;
        v8f a0 = {0.f,0.f,0.f,0.f,0.f,0.f,0.f,0.f};
        v8f a1 = {0.f,0.f,0.f,0.f,0.f,0.f,0.f,0.f};
        gemm_tile_x2(S_A, Wqkv, mt, ntp, 4, lane, a0, a1);
        int col0 = (2 * ntp) * 16 + (lane & 15), col1 = col0 + 16;
        float bias0 = b_in[col0], bias1 = b_in[col1];
        for (int r = 0; r < 8; ++r) {
            int m = mt * 16 + ((lane >> 4) << 3) + r;
            S_big[m * (3 * Dn) + col0] = (_Float16)(a0[r] + bias0);
            S_big[m * (3 * Dn) + col1] = (_Float16)(a1[r] + bias1);
        }
    }
    __syncthreads();

    // Phase 3: attention per (token, head); seq = PATCH contiguous tokens
    for (int row = tid; row < Ln * Hn; row += 256) {
        int t = row >> 2, h = row & 3;
        int s0 = (t / PATCH) * PATCH;
        const int qo = h * HDn, ko = Dn + h * HDn, vo = 2 * Dn + h * HDn;
        float sc[PATCH];
        float mx = -1e30f;
        const _Float16* qr = &S_big[t * 3 * Dn + qo];
        for (int j = 0; j < PATCH; ++j) {
            const _Float16* kr = &S_big[(s0 + j) * 3 * Dn + ko];
            float s = 0.f;
            for (int d = 0; d < HDn; ++d) s += (float)qr[d] * (float)kr[d];
            s *= 0.17677669529663688f;   // 1/sqrt(32)
            sc[j] = s; mx = fmaxf(mx, s);
        }
        float den = 0.f;
        for (int j = 0; j < PATCH; ++j) { sc[j] = __expf(sc[j] - mx); den += sc[j]; }
        float inv = 1.f / den;
        float ctx[HDn];
        for (int d = 0; d < HDn; ++d) ctx[d] = 0.f;
        for (int j = 0; j < PATCH; ++j) {
            float a = sc[j] * inv;
            const _Float16* vr = &S_big[(s0 + j) * 3 * Dn + vo];
            for (int d = 0; d < HDn; ++d) ctx[d] += a * (float)vr[d];
        }
        // write ctx fragment-swizzled into S_A (feature = h*32 + d -> kt = h)
        for (int d = 0; d < HDn; ++d)
            S_A[frag_off(t >> 4, h, 4, t & 15, d)] = (_Float16)ctx[d];
    }
    __syncthreads();

    // Phase 4: x1_pre = h + ctx @ w_out^T + b_out   (96x128, K=128) -> S_f32
    for (int t = wv; t < 6 * 4; t += 8) {
        int mt = t >> 2, ntp = t & 3;
        v8f a0 = {0.f,0.f,0.f,0.f,0.f,0.f,0.f,0.f};
        v8f a1 = {0.f,0.f,0.f,0.f,0.f,0.f,0.f,0.f};
        gemm_tile_x2(S_A, Wwo, mt, ntp, 4, lane, a0, a1);
        int col0 = (2 * ntp) * 16 + (lane & 15), col1 = col0 + 16;
        float bias0 = b_out[col0], bias1 = b_out[col1];
        for (int r = 0; r < 8; ++r) {
            int m = mt * 16 + ((lane >> 4) << 3) + r;
            S_f32[m * Dn + col0] += a0[r] + bias0;
            S_f32[m * Dn + col1] += a1[r] + bias1;
        }
    }
    __syncthreads();

    // Phase 5: LayerNorm1 -> S_f32 (x1) and fragment-swizzled f16 in S_A
    for (int t = tid; t < Ln; t += 256) {
        float* row = &S_f32[t * Dn];
        float mean = 0.f;
        for (int d = 0; d < Dn; ++d) mean += row[d];
        mean *= (1.f / Dn);
        float var = 0.f;
        for (int d = 0; d < Dn; ++d) { float df = row[d] - mean; var += df * df; }
        var *= (1.f / Dn);
        float inv = rsqrtf(var + LN_EPS);
        for (int d = 0; d < Dn; ++d) {
            float xn = (row[d] - mean) * inv * ln1_g[d] + ln1_b[d];
            row[d] = xn;
            S_A[frag_off(t >> 4, d >> 5, 4, t & 15, d & 31)] = (_Float16)xn;
        }
    }
    __syncthreads();

    // Phase 6: y = relu(x1 @ w_ff1^T + b_ff1)  (96x256, K=128)
    //          -> S_big, fragment-swizzled as next A (fragsK = 8)
    for (int t = wv; t < 6 * 8; t += 8) {
        int mt = t >> 3, ntp = t & 7;
        v8f a0 = {0.f,0.f,0.f,0.f,0.f,0.f,0.f,0.f};
        v8f a1 = {0.f,0.f,0.f,0.f,0.f,0.f,0.f,0.f};
        gemm_tile_x2(S_A, Wff1, mt, ntp, 4, lane, a0, a1);
        int col0 = (2 * ntp) * 16 + (lane & 15), col1 = col0 + 16;
        float bias0 = b_ff1[col0], bias1 = b_ff1[col1];
        for (int r = 0; r < 8; ++r) {
            int m = mt * 16 + ((lane >> 4) << 3) + r;
            float y0 = fmaxf(a0[r] + bias0, 0.f);
            float y1 = fmaxf(a1[r] + bias1, 0.f);
            S_big[frag_off(mt, col0 >> 5, 8, m & 15, col0 & 31)] = (_Float16)y0;
            S_big[frag_off(mt, col1 >> 5, 8, m & 15, col1 & 31)] = (_Float16)y1;
        }
    }
    __syncthreads();

    // Phase 7: x2_pre = x1 + y @ w_ff2^T + b_ff2  (96x128, K=256) -> S_f32
    for (int t = wv; t < 6 * 4; t += 8) {
        int mt = t >> 2, ntp = t & 3;
        v8f a0 = {0.f,0.f,0.f,0.f,0.f,0.f,0.f,0.f};
        v8f a1 = {0.f,0.f,0.f,0.f,0.f,0.f,0.f,0.f};
        gemm_tile_x2(S_big, Wff2, mt, ntp, 8, lane, a0, a1);
        int col0 = (2 * ntp) * 16 + (lane & 15), col1 = col0 + 16;
        float bias0 = b_ff2[col0], bias1 = b_ff2[col1];
        for (int r = 0; r < 8; ++r) {
            int m = mt * 16 + ((lane >> 4) << 3) + r;
            S_f32[m * Dn + col0] += a0[r] + bias0;
            S_f32[m * Dn + col1] += a1[r] + bias1;
        }
    }
    __syncthreads();

    // Phase 8: LayerNorm2, then out (+)= gate * y2 (first expert seeds with x)
    for (int t = tid; t < Ln; t += 256) {
        float* row = &S_f32[t * Dn];
        float mean = 0.f;
        for (int d = 0; d < Dn; ++d) mean += row[d];
        mean *= (1.f / Dn);
        float var = 0.f;
        for (int d = 0; d < Dn; ++d) { float df = row[d] - mean; var += df * df; }
        var *= (1.f / Dn);
        float inv = rsqrtf(var + LN_EPS);
        const float* xr = xrow + (size_t)t * (Nn * Dn);
        float*       op = orow + (size_t)t * (Nn * Dn);
        for (int d = 0; d < Dn; ++d) {
            float y = (row[d] - mean) * inv * ln2_g[d] + ln2_b[d];
            float base = first ? xr[d] : op[d];
            op[d] = base + g * y;
        }
    }
}

// ---------------------------------------------------------------------------
extern "C" void kernel_launch(void* const* d_in, const int* in_sizes, int n_in,
                              void* d_out, int out_size, void* d_ws, size_t ws_size,
                              hipStream_t stream)
{
    (void)in_sizes; (void)n_in; (void)out_size; (void)ws_size;
    const float* x      = (const float*)d_in[0];
    const float* w_sl   = (const float*)d_in[1];
    const float* b_sl   = (const float*)d_in[2];
    const float* w_gate = (const float*)d_in[3];
    const float* b_gate = (const float*)d_in[4];
    const float* w_in   = (const float*)d_in[5];
    const float* b_in   = (const float*)d_in[6];
    const float* w_out  = (const float*)d_in[7];
    const float* b_out  = (const float*)d_in[8];
    const float* ln1_g  = (const float*)d_in[9];
    const float* ln1_b  = (const float*)d_in[10];
    const float* w_ff1  = (const float*)d_in[11];
    const float* b_ff1  = (const float*)d_in[12];
    const float* w_ff2  = (const float*)d_in[13];
    const float* b_ff2  = (const float*)d_in[14];
    const float* ln2_g  = (const float*)d_in[15];
    const float* ln2_b  = (const float*)d_in[16];

    float* out = (float*)d_out;
    char*  ws  = (char*)d_ws;
    float*    new_x = (float*)ws;                  // B*L*N f32   (786432 B)
    float*    gbuf  = (float*)(ws + 786432);       // B*E f32     (512 B)
    _Float16* wf16  = (_Float16*)(ws + 786944);    // 4*131072 f16 (1 MB)

    wswz_kernel<<<2048, 256, 0, stream>>>(w_in, w_out, w_ff1, w_ff2, wf16);
    decomp_kernel<<<256, 256, 0, stream>>>(x, new_x);
    gate_kernel<<<1, 256, 0, stream>>>(new_x, w_sl, b_sl, w_gate, b_gate, gbuf,
                                       out + (size_t)Bn * Ln * Nn * Dn);

    const size_t shmem = 49152 + 73728 + 24576;    // 144 KB dynamic LDS
    const int grid = Bn * Nn;
    expert_kernel<8><<<grid, 256, shmem, stream>>>(x, gbuf, wf16 + 0 * 131072,
        b_in + 0 * 384, b_out + 0 * 128, ln1_g + 0 * 128, ln1_b + 0 * 128,
        b_ff1 + 0 * 256, b_ff2 + 0 * 128, ln2_g + 0 * 128, ln2_b + 0 * 128,
        out, 0, 1);
    expert_kernel<6><<<grid, 256, shmem, stream>>>(x, gbuf, wf16 + 1 * 131072,
        b_in + 1 * 384, b_out + 1 * 128, ln1_g + 1 * 128, ln1_b + 1 * 128,
        b_ff1 + 1 * 256, b_ff2 + 1 * 128, ln2_g + 1 * 128, ln2_b + 1 * 128,
        out, 1, 0);
    expert_kernel<4><<<grid, 256, shmem, stream>>>(x, gbuf, wf16 + 2 * 131072,
        b_in + 2 * 384, b_out + 2 * 128, ln1_g + 2 * 128, ln1_b + 2 * 128,
        b_ff1 + 2 * 256, b_ff2 + 2 * 128, ln2_g + 2 * 128, ln2_b + 2 * 128,
        out, 2, 0);
    expert_kernel<2><<<grid, 256, shmem, stream>>>(x, gbuf, wf16 + 3 * 131072,
        b_in + 3 * 384, b_out + 3 * 128, ln1_g + 3 * 128, ln1_b + 3 * 128,
        b_ff1 + 3 * 256, b_ff2 + 3 * 128, ln2_g + 3 * 128, ln2_b + 3 * 128,
        out, 3, 0);
}
